// ALiBiAttention_64914135712467
// MI455X (gfx1250) — matrix operand order repaired and checked by reading, not compiled, not run on hardware
//
#include <hip/hip_runtime.h>

// ALiBi attention for MI455X (gfx1250, wave32, WMMA + async global->LDS DMA).
// B=8, L=1024, H=8, E=64, fp32 I/O. Outputs V_out [B,L,H,E] then series [B,H,L,L].
//
// Roofline: mandatory HBM traffic ~268MB series write + ~48MB reads -> ~14us @ 23.3TB/s.
// GEMMs on v_wmma_f32_16x16x32_f16. K/V staged via GLOBAL_LOAD_ASYNC_TO_LDS_B64 with
// double buffering (ASYNCcnt), so global latency hides behind the WMMA tile loop even
// at 1 block/WGP occupancy.

typedef __attribute__((ext_vector_type(16))) _Float16 v16h;
typedef __attribute__((ext_vector_type(8)))  float    v8f;

#define LSEQ 1024
#define NH   8
#define NE   64
#define NB   8
#define ROWF (NH * NE)        // 512 floats between consecutive seq positions
#define SROW 1028             // score-panel row stride: 1024+4 pad (bank de-conflict)
#define TROW 70               // staging row stride (floats): 6m mod 64 distinct, 8B aligned
#define CH   64               // staging chunk rows
#define STGSZ (CH * TROW)     // floats per staging buffer (17,920 B)
#define SOFF (64 * SROW)      // float offset of staging buffers in LDS

// One 64-row x 64-float chunk -> LDS buffer via async DMA.
// Each thread issues 16 x b64 transfers => 16 per wave => ASYNCcnt += 16 per wave.
__device__ __forceinline__ void stage_chunk_async(const float* __restrict__ gbase,
                                                  float* buf) {
  const int t = threadIdx.x;
#pragma unroll
  for (int k = 0; k < 16; ++k) {
    const int g   = t + 128 * k;   // 0..2047 : 64 rows x 32 8-byte segments
    const int row = g >> 5;
    const int s8  = g & 31;
    uint32_t lds_addr = (uint32_t)(uintptr_t)(buf + row * TROW + s8 * 2);
    const float* gaddr = gbase + (size_t)row * ROWF + s8 * 2;
    asm volatile("global_load_async_to_lds_b64 %0, %1, off"
                 :: "v"(lds_addr), "v"(gaddr) : "memory");
  }
}

__launch_bounds__(128, 1)
__global__ void alibi_attn_kernel(const float* __restrict__ Q,
                                  const float* __restrict__ K,
                                  const float* __restrict__ V,
                                  float* __restrict__ Out,      // [B,L,H,E]
                                  float* __restrict__ Series) { // [B,H,L,L]
  extern __shared__ float S[];          // [64 x SROW] scores | 2 x [CH x TROW] stage
  float* STG0 = S + SOFF;
  float* STG1 = STG0 + STGSZ;

  const int lane = threadIdx.x & 31;
  const int wv   = threadIdx.x >> 5;    // wave 0..3, owns query rows [16*wv, 16*wv+16)
  const int m    = lane & 15;           // A-row / B-column index within 16
  const int half = lane >> 4;           // lane half selects K-chunk per ISA layout

  const int qblk  = blockIdx.x & 15;
  const int bh    = blockIdx.x >> 4;
  const int h     = bh & 7;
  const int b     = bh >> 3;
  const int qbase = qblk * 64;

  const float scale = 0.125f;                               // 1/sqrt(64)
  const float slope = exp2f(-(float)(h + 1) * 0.125f);      // ALiBi: 2^{-(h+1)/8}

  const float* Qb = Q + (size_t)b * LSEQ * ROWF + h * NE;
  const float* Kb = K + (size_t)b * LSEQ * ROWF + h * NE;
  const float* Vb = V + (size_t)b * LSEQ * ROWF + h * NE;

  // ---- Q A-fragments (16x32 f16), loop-invariant: lane half 0 holds K {0..7,16..23} ----
  const float* Qrow = Qb + (size_t)(qbase + wv * 16 + m) * ROWF;
  v16h aQ0, aQ1;
#pragma unroll
  for (int i = 0; i < 8; ++i) {
    aQ0[i]     = (_Float16)Qrow[half * 8 + i];
    aQ0[8 + i] = (_Float16)Qrow[16 + half * 8 + i];
    aQ1[i]     = (_Float16)Qrow[32 + half * 8 + i];
    aQ1[8 + i] = (_Float16)Qrow[48 + half * 8 + i];
  }

  // ================= Phase 1: S = scale * Q K^T + alibi =================
  stage_chunk_async(Kb, STG0);                       // prefetch chunk 0
  for (int c = 0; c < 16; ++c) {
    float* cur = (c & 1) ? STG1 : STG0;
    if (c + 1 < 16) {                                // prefetch next chunk, overlap compute
      stage_chunk_async(Kb + (size_t)(c + 1) * CH * ROWF, (c & 1) ? STG0 : STG1);
      asm volatile("s_wait_asynccnt 0x10" ::: "memory");  // only current 16 done (in-order)
    } else {
      asm volatile("s_wait_asynccnt 0x0" ::: "memory");
    }
    __syncthreads();
    for (int tt = 0; tt < 4; ++tt) {
      // B-frag (32x16 f16) from LDS: lane = column n; half 0 holds K {0..7,16..23}, half 1: {8..15,24..31}
      const float* Krow = cur + (tt * 16 + m) * TROW;
      v16h bK0, bK1;
#pragma unroll
      for (int j = 0; j < 16; j += 2) {
        float2 x0 = *(const float2*)(Krow + half * 8 + j + (j & 8));
        float2 x1 = *(const float2*)(Krow + 32 + half * 8 + j + (j & 8));
        bK0[j] = (_Float16)x0.x; bK0[j + 1] = (_Float16)x0.y;
        bK1[j] = (_Float16)x1.x; bK1[j + 1] = (_Float16)x1.y;
      }
      v8f acc = {};
      acc = __builtin_amdgcn_wmma_f32_16x16x32_f16(false, aQ0, false, bK0,
                                                   (short)0, acc, false, false);
      acc = __builtin_amdgcn_wmma_f32_16x16x32_f16(false, aQ1, false, bK1,
                                                   (short)0, acc, false, false);
      // C layout: VGPR j -> row (j + 8*half), column = lane%16
      const int   col  = c * CH + tt * 16 + m;
      const float bias = slope * ((float)col - 1023.0f);
      float* srow = &S[(size_t)(wv * 16 + 8 * half) * SROW + col];
#pragma unroll
      for (int j = 0; j < 8; ++j)
        srow[(size_t)j * SROW] = acc[j] * scale + bias;
    }
    __syncthreads();
  }

  // ================= Phase 2: row softmax (float4 wide); stream series =================
  const float LOG2E = 1.44269504088896f;
  for (int rl = 0; rl < 16; ++rl) {
    const int row = wv * 16 + rl;
    float4* S4 = (float4*)&S[(size_t)row * SROW];   // 256 float4s, 16B-aligned (4112*row)

    float mx = -3.402823466e38f;
    for (int c = lane; c < 256; c += 32) {
      float4 v = S4[c];
      mx = fmaxf(mx, fmaxf(fmaxf(v.x, v.y), fmaxf(v.z, v.w)));
    }
#pragma unroll
    for (int off = 16; off > 0; off >>= 1) mx = fmaxf(mx, __shfl_xor(mx, off, 32));

    float sum = 0.f;
    for (int c = lane; c < 256; c += 32) {
      float4 v = S4[c];
      v.x = exp2f((v.x - mx) * LOG2E);
      v.y = exp2f((v.y - mx) * LOG2E);
      v.z = exp2f((v.z - mx) * LOG2E);
      v.w = exp2f((v.w - mx) * LOG2E);
      S4[c] = v;
      sum += (v.x + v.y) + (v.z + v.w);
    }
#pragma unroll
    for (int off = 16; off > 0; off >>= 1) sum += __shfl_xor(sum, off, 32);

    const float inv = 1.0f / sum;
    float4* ser4 = (float4*)(Series +
                   ((size_t)(b * NH + h) * LSEQ + (qbase + row)) * LSEQ);
    for (int c = lane; c < 256; c += 32) {
      float4 v = S4[c];
      v.x *= inv; v.y *= inv; v.z *= inv; v.w *= inv;
      S4[c]   = v;        // kept as P for phase 3
      ser4[c] = v;        // b128 stores, 512B/wave/instr -> HBM-bound path
    }
  }
  // No barrier needed: S rows are wave-private; STG is idle during phase 2.

  // ================= Phase 3: O = P * V =================
  v8f acc0 = {}, acc1 = {}, acc2 = {}, acc3 = {};
  const float* Prow = &S[(size_t)(wv * 16 + m) * SROW];
  stage_chunk_async(Vb, STG0);                       // prefetch V chunk 0
  for (int c = 0; c < 16; ++c) {
    const float* cur = (c & 1) ? STG1 : STG0;
    if (c + 1 < 16) {
      stage_chunk_async(Vb + (size_t)(c + 1) * CH * ROWF, (c & 1) ? STG0 : STG1);
      asm volatile("s_wait_asynccnt 0x10" ::: "memory");
    } else {
      asm volatile("s_wait_asynccnt 0x0" ::: "memory");
    }
    __syncthreads();
    for (int ss = 0; ss < CH; ss += 32) {
      const int s0 = c * CH + ss;
      // A-frag = P chunk 16x32, aligned float4 LDS reads
      float4 p0 = *(const float4*)(Prow + s0 + half * 8);
      float4 p1 = *(const float4*)(Prow + s0 + half * 8 + 4);
      float4 p2 = *(const float4*)(Prow + s0 + 16 + half * 8);
      float4 p3 = *(const float4*)(Prow + s0 + 16 + half * 8 + 4);
      v16h aP;
      aP[0]  = (_Float16)p0.x; aP[1]  = (_Float16)p0.y;
      aP[2]  = (_Float16)p0.z; aP[3]  = (_Float16)p0.w;
      aP[4]  = (_Float16)p1.x; aP[5]  = (_Float16)p1.y;
      aP[6]  = (_Float16)p1.z; aP[7]  = (_Float16)p1.w;
      aP[8]  = (_Float16)p2.x; aP[9]  = (_Float16)p2.y;
      aP[10] = (_Float16)p2.z; aP[11] = (_Float16)p2.w;
      aP[12] = (_Float16)p3.x; aP[13] = (_Float16)p3.y;
      aP[14] = (_Float16)p3.z; aP[15] = (_Float16)p3.w;
      // B-frags = V chunk 32x64 from LDS (conflict-free column gather: halves 48 banks apart)
      v16h b0, b1, b2, b3;
#pragma unroll
      for (int j = 0; j < 16; ++j) {
        const float* Vr = cur + (ss + half * 8 + j + (j & 8)) * TROW;
        b0[j] = (_Float16)Vr[m];
        b1[j] = (_Float16)Vr[16 + m];
        b2[j] = (_Float16)Vr[32 + m];
        b3[j] = (_Float16)Vr[48 + m];
      }
      acc0 = __builtin_amdgcn_wmma_f32_16x16x32_f16(false, aP, false, b0,
                                                    (short)0, acc0, false, false);
      acc1 = __builtin_amdgcn_wmma_f32_16x16x32_f16(false, aP, false, b1,
                                                    (short)0, acc1, false, false);
      acc2 = __builtin_amdgcn_wmma_f32_16x16x32_f16(false, aP, false, b2,
                                                    (short)0, acc2, false, false);
      acc3 = __builtin_amdgcn_wmma_f32_16x16x32_f16(false, aP, false, b3,
                                                    (short)0, acc3, false, false);
    }
    __syncthreads();
  }

#pragma unroll
  for (int j = 0; j < 8; ++j) {
    const int qrow = qbase + wv * 16 + j + 8 * half;
    float* orow = Out + ((size_t)(b * LSEQ + qrow) * NH + h) * NE;
    orow[m]      = acc0[j];
    orow[16 + m] = acc1[j];
    orow[32 + m] = acc2[j];
    orow[48 + m] = acc3[j];
  }
}

extern "C" void kernel_launch(void* const* d_in, const int* in_sizes, int n_in,
                              void* d_out, int out_size, void* d_ws, size_t ws_size,
                              hipStream_t stream) {
  const float* Q = (const float*)d_in[0];
  const float* K = (const float*)d_in[1];
  const float* V = (const float*)d_in[2];
  // d_in[3] = patch_index (-1) -> identity branch, ignored.

  float* Out    = (float*)d_out;
  float* Series = Out + (size_t)NB * LSEQ * NH * NE;   // outputs concatenated flat

  const size_t shmem = (size_t)(SOFF + 2 * STGSZ) * sizeof(float);  // 299008 B < 320KB
  (void)hipFuncSetAttribute((const void*)alibi_attn_kernel,
                            hipFuncAttributeMaxDynamicSharedMemorySize,
                            (int)shmem);

  dim3 grid(NB * NH * (LSEQ / 64));   // 1024 blocks: one per (b, h, 64-row q-block)
  dim3 block(128);                    // 4 wave32 waves
  alibi_attn_kernel<<<grid, block, shmem, stream>>>(Q, K, V, Out, Series);
}
